// GPSTransformer_14963666059755
// MI455X (gfx1250) — compile-verified, matrix operand order checked by reading, not run on hardware
//
#include <hip/hip_runtime.h>
#include <hip/hip_bf16.h>
#include <math.h>

// ---------------- problem constants ----------------
#define NN    16384
#define BG    16
#define LG    1024
#define EE    262144
#define CC    128
#define PEC   16
#define INC   32
#define EDC   16
#define OUTC  32
#define NL    3
#define HH    4
#define DHH   32

typedef __attribute__((ext_vector_type(16))) _Float16 v16h;
typedef __attribute__((ext_vector_type(8)))  float    v8f;

__device__ __forceinline__ int frag_k(int v, int half) {
  // 16-bit A/B fragment K index for VGPR v (pairs), per CDNA5 ISA layout
  return (v < 4 ? 2 * v : 16 + 2 * (v - 4)) + 8 * half;
}

// convert 8 consecutive f32 -> 8 f16, store as one 16B LDS write
__device__ __forceinline__ void cvt_store8(_Float16* dst, const float* src) {
  float4 f0 = *(const float4*)src;
  float4 f1 = *(const float4*)(src + 4);
  union { _Float16 h[8]; uint4 u; } pk;
  pk.h[0] = (_Float16)f0.x; pk.h[1] = (_Float16)f0.y;
  pk.h[2] = (_Float16)f0.z; pk.h[3] = (_Float16)f0.w;
  pk.h[4] = (_Float16)f1.x; pk.h[5] = (_Float16)f1.y;
  pk.h[6] = (_Float16)f1.z; pk.h[7] = (_Float16)f1.w;
  *(uint4*)dst = pk.u;
}

// ============================================================
// WMMA GEMM: C = act(A[M,K] @ W[K,N] + bias)
// Specialized: M % 128 == 0, K % 32 == 0 (true for every call here).
// act: 0=none, 1=leaky(0.01), 2=relu
// block: 256 threads = 8 waves (4 Mtiles x 2 Ntiles of 32x32 each),
// block tile 128M x 64N, 4 WMMAs per wave per K-step.
// ============================================================
__global__ __launch_bounds__(256) void gemm_wmma_kernel(
    const float* __restrict__ A, const float* __restrict__ W,
    const float* __restrict__ bias, float* __restrict__ Cout,
    int M, int Nn, int K, int act) {
  __shared__ __attribute__((aligned(16))) _Float16 As[128 * 32]; // [m][k]
  __shared__ __attribute__((aligned(16))) _Float16 Bs[64 * 32];  // [n][k] transposed
  int tid  = threadIdx.x;
  int wave = tid >> 5, lane = tid & 31;
  int wm = wave & 3, wn = wave >> 2;     // wm 0..3 (M), wn 0..1 (N)
  int half = lane >> 4, r = lane & 15;
  int mblk = blockIdx.y * 128;
  int nblk = blockIdx.x * 64;

  v8f c00 = {}, c01 = {}, c10 = {}, c11 = {};

  // A staging: thread -> (row = tid/2, k-half = (tid&1)*16); 16 f16 per thread
  int sa_m = tid >> 1;
  int sa_k = (tid & 1) * 16;
  const float* Aptr = A + (size_t)(mblk + sa_m) * K + sa_k;
  _Float16* As_dst = &As[sa_m * 32 + sa_k];

  // B staging: thread -> (n = tid&63, kbase = (tid>>6)*8); 8 f16 per thread.
  // Out-of-range columns are CLAMPED (always a legal address) and the value
  // masked to zero -> unconditional global loads, no exec-mask branching.
  int sb_n = tid & 63;
  int sb_k = (tid >> 6) * 8;   // 0,8,16,24
  int gn_b = nblk + sb_n;
  float bmask = (gn_b < Nn) ? 1.f : 0.f;
  int gn_bc = (gn_b < Nn) ? gn_b : (Nn - 1);
  const float* Wptr = W + (size_t)sb_k * Nn + gn_bc;
  _Float16* Bs_dst = &Bs[sb_n * 32 + sb_k];

  for (int k0 = 0; k0 < K; k0 += 32) {
    __syncthreads();
    // ---- stage A tile (vectorized b128 loads, packed f16 b128 stores) ----
    cvt_store8(As_dst,     Aptr + k0);
    cvt_store8(As_dst + 8, Aptr + k0 + 8);
    if (k0 + 32 < K) __builtin_prefetch(Aptr + k0 + 32, 0, 0);
    // ---- stage B tile transposed (branch-free) ----
    {
      union { _Float16 h[8]; uint4 u; } pk;
#pragma unroll
      for (int j = 0; j < 8; ++j) {
        float v = Wptr[(size_t)(k0 + j) * Nn] * bmask;
        pk.h[j] = (_Float16)v;
      }
      *(uint4*)Bs_dst = pk.u;
    }
    __syncthreads();
    // ---- fragments (pair-contiguous: compiler merges to ds_load_b128) ----
    union { unsigned u[8]; v16h h; } a0, a1, b0, b1;
#pragma unroll
    for (int v = 0; v < 8; ++v) {
      int kk = frag_k(v, half);
      a0.u[v] = *(const unsigned*)&As[(wm * 32 + r) * 32 + kk];
      a1.u[v] = *(const unsigned*)&As[(wm * 32 + 16 + r) * 32 + kk];
      b0.u[v] = *(const unsigned*)&Bs[(wn * 32 + r) * 32 + kk];
      b1.u[v] = *(const unsigned*)&Bs[(wn * 32 + 16 + r) * 32 + kk];
    }
    c00 = __builtin_amdgcn_wmma_f32_16x16x32_f16(false, a0.h, false, b0.h, (short)0, c00, false, false);
    c01 = __builtin_amdgcn_wmma_f32_16x16x32_f16(false, a0.h, false, b1.h, (short)0, c01, false, false);
    c10 = __builtin_amdgcn_wmma_f32_16x16x32_f16(false, a1.h, false, b0.h, (short)0, c10, false, false);
    c11 = __builtin_amdgcn_wmma_f32_16x16x32_f16(false, a1.h, false, b1.h, (short)0, c11, false, false);
  }

  // ---- epilogue: bias + activation + guarded store (only N needs guards) ----
  int gn0 = nblk + wn * 32 + r;
  int gn1 = gn0 + 16;
  float bv0 = (gn0 < Nn) ? bias[gn0] : 0.f;
  float bv1 = (gn1 < Nn) ? bias[gn1] : 0.f;
#pragma unroll
  for (int v = 0; v < 8; ++v) {
    int row0 = mblk + wm * 32 + v + 8 * half;
    int row1 = row0 + 16;
    float v00 = c00[v] + bv0, v01 = c01[v] + bv1;
    float v10 = c10[v] + bv0, v11 = c11[v] + bv1;
    if (act == 1) {
      v00 = v00 >= 0.f ? v00 : 0.01f * v00; v01 = v01 >= 0.f ? v01 : 0.01f * v01;
      v10 = v10 >= 0.f ? v10 : 0.01f * v10; v11 = v11 >= 0.f ? v11 : 0.01f * v11;
    } else if (act == 2) {
      v00 = fmaxf(v00, 0.f); v01 = fmaxf(v01, 0.f);
      v10 = fmaxf(v10, 0.f); v11 = fmaxf(v11, 0.f);
    }
    if (gn0 < Nn) {
      Cout[(size_t)row0 * Nn + gn0] = v00;
      Cout[(size_t)row1 * Nn + gn0] = v10;
    }
    if (gn1 < Nn) {
      Cout[(size_t)row0 * Nn + gn1] = v01;
      Cout[(size_t)row1 * Nn + gn1] = v11;
    }
  }
}

// ============================================================
// Flash-style multi-head attention over equal-size graphs.
// qkv: [N, 384] (q | k | v).  out: [N, 128] (pre-Wo).
// block = 128 threads (4 waves), one (graph, head, 64-query chunk) per block.
// ============================================================
__global__ __launch_bounds__(128) void attn_kernel(
    const float* __restrict__ qkv, float* __restrict__ out) {
  int idx = blockIdx.x;
  int qchunk = idx & 15; idx >>= 4;   // LG/64 = 16 chunks
  int head = idx & 3;    idx >>= 2;   // H = 4
  int g = idx;                        // graph
  int wave = threadIdx.x >> 5;
  int lane = threadIdx.x & 31;
  int half = lane >> 4, r = lane & 15;
  const float scale = 0.17677669529663687f; // 1/sqrt(32)

  __shared__ __attribute__((aligned(16))) _Float16 kt[32 * 32];    // [key][d]
  __shared__ __attribute__((aligned(16))) _Float16 vt[32 * 32];    // [d][key]
  __shared__ __attribute__((aligned(16))) _Float16 qs[4][16 * 32]; // [q][d] scaled
  __shared__ __attribute__((aligned(16))) _Float16 pb[4][16 * 32]; // P tile f16
  __shared__ float sc[4][16 * 32];
  __shared__ float alphaS[4][16];
  __shared__ float lsumS[4][16];

  int qbase = g * LG + qchunk * 64 + wave * 16;
  for (int i = lane; i < 512; i += 32) {
    int qr = i >> 5, d = i & 31;
    qs[wave][i] = (_Float16)(qkv[(size_t)(qbase + qr) * 384 + head * 32 + d] * scale);
  }
  __syncthreads();
  union { unsigned u[8]; v16h h; } qa;
#pragma unroll
  for (int v = 0; v < 8; ++v)
    qa.u[v] = *(const unsigned*)&qs[wave][r * 32 + frag_k(v, half)];

  v8f o0 = {}, o1 = {};
  float mrun = -INFINITY, lrun = 0.f;   // row stats owned by lanes 0..15 (row=r)

  for (int t = 0; t < LG / 32; ++t) {
    __syncthreads();
    int kb = g * LG + t * 32;
    for (int i = threadIdx.x; i < 1024; i += 128) {
      int key = i >> 5, d = i & 31;
      kt[key * 32 + d] = (_Float16)qkv[(size_t)(kb + key) * 384 + 128 + head * 32 + d];
      vt[d * 32 + key] = (_Float16)qkv[(size_t)(kb + key) * 384 + 256 + head * 32 + d];
    }
    __syncthreads();
    union { unsigned u[8]; v16h h; } kb0, kb1;
#pragma unroll
    for (int v = 0; v < 8; ++v) {
      int kk = frag_k(v, half);
      kb0.u[v] = *(const unsigned*)&kt[r * 32 + kk];        // keys 0..15
      kb1.u[v] = *(const unsigned*)&kt[(16 + r) * 32 + kk]; // keys 16..31
    }
    v8f s0 = {}, s1 = {};
    s0 = __builtin_amdgcn_wmma_f32_16x16x32_f16(false, qa.h, false, kb0.h, (short)0, s0, false, false);
    s1 = __builtin_amdgcn_wmma_f32_16x16x32_f16(false, qa.h, false, kb1.h, (short)0, s1, false, false);
#pragma unroll
    for (int v = 0; v < 8; ++v) {
      int row = v + 8 * half;
      sc[wave][row * 32 + r]      = s0[v];
      sc[wave][row * 32 + 16 + r] = s1[v];
    }
    __syncthreads();
    if (half == 0) {  // lane r owns row r
      float tm = -INFINITY;
      for (int j = 0; j < 32; ++j) tm = fmaxf(tm, sc[wave][r * 32 + j]);
      float mnew = fmaxf(mrun, tm);
      float a = expf(mrun - mnew);
      float ts = 0.f;
      for (int j = 0; j < 32; ++j) {
        float p = expf(sc[wave][r * 32 + j] - mnew);
        pb[wave][r * 32 + j] = (_Float16)p;
        ts += p;
      }
      lrun = lrun * a + ts;
      mrun = mnew;
      alphaS[wave][r] = a;
    }
    __syncthreads();
    union { unsigned u[8]; v16h h; } pa, vb0, vb1;
#pragma unroll
    for (int v = 0; v < 8; ++v) {
      int kk = frag_k(v, half);
      pa.u[v]  = *(const unsigned*)&pb[wave][r * 32 + kk];
      vb0.u[v] = *(const unsigned*)&vt[r * 32 + kk];        // d = 0..15
      vb1.u[v] = *(const unsigned*)&vt[(16 + r) * 32 + kk]; // d = 16..31
    }
#pragma unroll
    for (int v = 0; v < 8; ++v) {
      float a = alphaS[wave][v + 8 * half];
      o0[v] *= a; o1[v] *= a;
    }
    o0 = __builtin_amdgcn_wmma_f32_16x16x32_f16(false, pa.h, false, vb0.h, (short)0, o0, false, false);
    o1 = __builtin_amdgcn_wmma_f32_16x16x32_f16(false, pa.h, false, vb1.h, (short)0, o1, false, false);
  }
  if (half == 0) lsumS[wave][r] = lrun;
  __syncthreads();
#pragma unroll
  for (int v = 0; v < 8; ++v) {
    int row = v + 8 * half;
    float inv = 1.f / lsumS[wave][row];
    size_t base = (size_t)(qbase + row) * CC + head * 32;
    out[base + r]      = o0[v] * inv;
    out[base + 16 + r] = o1[v] * inv;
  }
}

// ============================================================
// GINEConv edge kernel: agg[dst] += relu(h[src] + ea@We + be)
// 64 edges per block, thread = (edge, channel)
// ============================================================
__global__ __launch_bounds__(256) void gine_kernel(
    const float* __restrict__ h, const float* __restrict__ ea,
    const int* __restrict__ src, const int* __restrict__ dst,
    const float* __restrict__ We, const float* __restrict__ be,
    float* __restrict__ agg) {
  __shared__ float sWe[EDC * CC];
  __shared__ float sbe[CC];
  __shared__ float sea[64 * EDC];
  __shared__ int   ssrc[64], sdst[64];
  int tid = threadIdx.x;
  for (int i = tid; i < EDC * CC; i += 256) sWe[i] = We[i];
  if (tid < CC) sbe[tid] = be[tid];
  int ebase = blockIdx.x * 64;
  for (int i = tid; i < 64 * EDC; i += 256) sea[i] = ea[(size_t)ebase * EDC + i];
  if (tid < 64) { ssrc[tid] = src[ebase + tid]; sdst[tid] = dst[ebase + tid]; }
  __syncthreads();
  int c = tid & (CC - 1);
  for (int i = tid >> 7; i < 64; i += 2) {
    int s = ssrc[i], d = sdst[i];
    float acc = sbe[c] + h[(size_t)s * CC + c];
#pragma unroll
    for (int k = 0; k < EDC; ++k) acc += sea[i * EDC + k] * sWe[k * CC + c];
    acc = acc > 0.f ? acc : 0.f;
    atomicAdd(&agg[(size_t)d * CC + c], acc);
  }
}

// ============================================================
// Column-wise training-mode BatchNorm (one block per column),
// optional fused residual input, strided/offset output (for concat).
// ============================================================
__global__ __launch_bounds__(256) void bn_kernel(
    const float* __restrict__ in, const float* __restrict__ res,
    const float* __restrict__ g, const float* __restrict__ b,
    float* __restrict__ out, int rows, int cols, int ostride, int coff) {
  int c = blockIdx.x;
  int tid = threadIdx.x;
  __shared__ float s1[256], s2[256];
  float sum = 0.f, sq = 0.f;
  for (int rr = tid; rr < rows; rr += 256) {
    float v = in[(size_t)rr * cols + c];
    if (res) v += res[(size_t)rr * cols + c];
    sum += v; sq += v * v;
  }
  s1[tid] = sum; s2[tid] = sq;
  __syncthreads();
  for (int s = 128; s > 0; s >>= 1) {
    if (tid < s) { s1[tid] += s1[tid + s]; s2[tid] += s2[tid + s]; }
    __syncthreads();
  }
  float mean = s1[0] / rows;
  float var  = s2[0] / rows - mean * mean;
  float rstd = rsqrtf(var + 1e-5f);
  float gg = g[c], bb = b[c];
  for (int rr = tid; rr < rows; rr += 256) {
    float v = in[(size_t)rr * cols + c];
    if (res) v += res[(size_t)rr * cols + c];
    out[(size_t)rr * ostride + coff + c] = (v - mean) * rstd * gg + bb;
  }
}

__global__ __launch_bounds__(256) void add_kernel(
    const float* __restrict__ a, const float* __restrict__ b,
    float* __restrict__ o, int n) {
  int i = blockIdx.x * 256 + threadIdx.x;
  if (i < n) o[i] = a[i] + b[i];
}

__global__ __launch_bounds__(256) void zero_kernel(float* __restrict__ o, int n) {
  int i = blockIdx.x * 256 + threadIdx.x;
  if (i < n) o[i] = 0.f;
}

// ============================================================
extern "C" void kernel_launch(void* const* d_in, const int* in_sizes, int n_in,
                              void* d_out, int out_size, void* d_ws, size_t ws_size,
                              hipStream_t stream) {
  const float* x     = (const float*)d_in[0];
  const float* pe    = (const float*)d_in[1];
  const int*   ei    = (const int*)d_in[2];
  const float* ea    = (const float*)d_in[3];
  const float* W_enc = (const float*)d_in[5];
  const float* b_enc = (const float*)d_in[6];
  const float* in_g  = (const float*)d_in[7];
  const float* in_b  = (const float*)d_in[8];
  const float* pe_g  = (const float*)d_in[9];
  const float* pe_b  = (const float*)d_in[10];
  const float* We    = (const float*)d_in[11];
  const float* be    = (const float*)d_in[12];
  const float* Wm    = (const float*)d_in[13];
  const float* bm    = (const float*)d_in[14];
  const float* Wqkv  = (const float*)d_in[15];
  const float* bqkv  = (const float*)d_in[16];
  const float* Wo    = (const float*)d_in[17];
  const float* bo    = (const float*)d_in[18];
  const float* W1    = (const float*)d_in[19];
  const float* b1    = (const float*)d_in[20];
  const float* W2    = (const float*)d_in[21];
  const float* b2    = (const float*)d_in[22];
  const float* n1g   = (const float*)d_in[23];
  const float* n1b   = (const float*)d_in[24];
  const float* n2g   = (const float*)d_in[25];
  const float* n2b   = (const float*)d_in[26];
  const float* n3g   = (const float*)d_in[27];
  const float* n3b   = (const float*)d_in[28];
  const float* nLg   = (const float*)d_in[29];
  const float* nLb   = (const float*)d_in[30];
  const float* pd_g  = (const float*)d_in[31];
  const float* pd_b  = (const float*)d_in[32];
  const float* Wd1   = (const float*)d_in[33];
  const float* bd1   = (const float*)d_in[34];
  const float* Wd2   = (const float*)d_in[35];
  const float* bd2   = (const float*)d_in[36];

  const int* srcI = ei;
  const int* dstI = ei + EE;

  // workspace layout
  float* ws = (float*)d_ws;
  size_t off = 0;
  float* h     = ws + off; off += (size_t)NN * CC;
  float* agg   = ws + off; off += (size_t)NN * CC;
  float* tA    = ws + off; off += (size_t)NN * CC;
  float* tB    = ws + off; off += (size_t)NN * CC;
  float* local = ws + off; off += (size_t)NN * CC;
  float* qkv   = ws + off; off += (size_t)NN * 3 * CC;
  float* attnO = ws + off; off += (size_t)NN * CC;
  float* attn  = ws + off; off += (size_t)NN * CC;
  float* out0  = ws + off; off += (size_t)NN * CC;
  float* ffn1  = ws + off; off += (size_t)NN * 2 * CC;
  float* ffn2  = ws + off; off += (size_t)NN * CC;
  float* enc   = ws + off; off += (size_t)NN * (CC - PEC);

  const int nel = NN * CC;
  dim3 blk256(256);
  dim3 gEl((nel + 255) / 256);

  auto gemm = [&](const float* A, const float* W, const float* bias, float* C,
                  int M, int Nn, int K, int act) {
    dim3 grid((Nn + 63) / 64, M / 128);
    gemm_wmma_kernel<<<grid, blk256, 0, stream>>>(A, W, bias, C, M, Nn, K, act);
  };
  auto bn = [&](const float* in, const float* res, const float* g, const float* b,
                float* out, int cols, int ostride, int coff) {
    bn_kernel<<<dim3(cols), blk256, 0, stream>>>(in, res, g, b, out, NN, cols, ostride, coff);
  };

  // ---- encoder + input norms + concat ----
  gemm(x, W_enc, b_enc, enc, NN, CC - PEC, INC, /*leaky*/1);
  bn(enc, nullptr, in_g, in_b, h, CC - PEC, CC, 0);
  bn(pe,  nullptr, pe_g, pe_b, h, PEC, CC, CC - PEC);

  // ---- GPS layers ----
  for (int l = 0; l < NL; ++l) {
    // GINEConv
    zero_kernel<<<gEl, blk256, 0, stream>>>(agg, nel);
    gine_kernel<<<dim3(EE / 64), blk256, 0, stream>>>(
        h, ea, srcI, dstI, We + (size_t)l * EDC * CC, be + (size_t)l * CC, agg);
    add_kernel<<<gEl, blk256, 0, stream>>>(h, agg, tA, nel);
    gemm(tA, Wm + (size_t)l * CC * CC, bm + (size_t)l * CC, tB, NN, CC, CC, /*leaky*/1);
    bn(tB, h, n1g + (size_t)l * CC, n1b + (size_t)l * CC, local, CC, CC, 0);

    // global attention
    gemm(h, Wqkv + (size_t)l * CC * 3 * CC, bqkv + (size_t)l * 3 * CC, qkv, NN, 3 * CC, CC, 0);
    attn_kernel<<<dim3(BG * HH * (LG / 64)), dim3(128), 0, stream>>>(qkv, attnO);
    gemm(attnO, Wo + (size_t)l * CC * CC, bo + (size_t)l * CC, tB, NN, CC, CC, 0);
    bn(tB, h, n2g + (size_t)l * CC, n2b + (size_t)l * CC, attn, CC, CC, 0);

    // combine + FFN + norms
    add_kernel<<<gEl, blk256, 0, stream>>>(local, attn, out0, nel);
    gemm(out0, W1 + (size_t)l * CC * 2 * CC, b1 + (size_t)l * 2 * CC, ffn1, NN, 2 * CC, CC, /*relu*/2);
    gemm(ffn1, W2 + (size_t)l * 2 * CC * CC, b2 + (size_t)l * CC, ffn2, NN, CC, 2 * CC, 0);
    add_kernel<<<gEl, blk256, 0, stream>>>(out0, ffn2, tA, nel);
    bn(tA, nullptr, n3g + (size_t)l * CC, n3b + (size_t)l * CC, tB, CC, CC, 0);
    bn(tB, nullptr, nLg + (size_t)l * CC, nLb + (size_t)l * CC, h, CC, CC, 0);
  }

  // ---- decoder ----
  bn(h, nullptr, pd_g, pd_b, tA, CC, CC, 0);
  gemm(tA, Wd1, bd1, tB, NN, CC, CC, /*leaky*/1);
  gemm(tB, Wd2, bd2, (float*)d_out, NN, OUTC, CC, 0);
}
